// pRNN_th_47631187313126
// MI455X (gfx1250) — matrix operand order repaired and checked by reading, not compiled, not run on hardware
//
#include <hip/hip_runtime.h>
#include <stdint.h>

// ---------------------------------------------------------------------------
// pRNN_th forward for MI455X (gfx1250, wave32, WMMA).
//
//   y_t (9,4096,200) f32 | h_t (9,4096,1024) f32 | obs_target (9,4096,200) f32
//
// Pipeline (all on `stream`, serialized):
//   0) zero per-step barrier counters (determinism under graph replay)
//   1) pack W_hh/W_ih/W_out/x to bf16 (pad K: 203->224, readout rows 200->208)
//   2) xin = x @ W_ih^T           (WMMA bf16, f32 acc)
//   3) serial scan: 32 WGs, W_hh slices LDS-resident (bf16), L2 flag barrier
//   4) 8 phase GEMMs H_j = relu(H_{j-1} @ W_hh^T + b)  (WMMA, bf16 re-quant)
//   5) y = sigmoid(H_all @ W_out^T)                    (WMMA)
//   6) obs_target gather
//
// GEMM waves are register-blocked MR=4 in M: one B fragment feeds 4 WMMAs,
// cutting W_hh L2 re-reads 4x and raising WMMA density per vmem clause.
// ---------------------------------------------------------------------------

#define OBS_N   200
#define ACT_N   3
#define HID_N   1024
#define K_TH    8
#define NPH     9          // K_TH + 1 phases
#define T_LEN   4096
#define XK      224        // padded input K (203 -> 224, multiple of 32)
#define NOUTP   208        // padded readout cols (200 -> 208, multiple of 16)

#define SCAN_WGS     32
#define SCAN_ROWS    32    // rows of W_hh per workgroup
#define SCAN_THREADS 256   // 8 lanes per row

typedef __attribute__((ext_vector_type(16))) __bf16 v16bf;
typedef __attribute__((ext_vector_type(8)))  float  v8f;

__device__ __forceinline__ unsigned short f2bf(float f) {
  unsigned int u = __float_as_uint(f);
  u += 0x7FFFu + ((u >> 16) & 1u);        // round-to-nearest-even
  return (unsigned short)(u >> 16);
}

union FragU { v16bf v; uint4 q[2]; };

// A fragment: 16x32 bf16, row-major source. ISA layout: lanes 0-15 hold M=lane,
// K = {k0..k0+7, k0+16..k0+23}; lanes 16-31 hold K = {k0+8.., k0+24..}.
__device__ __forceinline__ v16bf load_a_frag(const unsigned short* __restrict__ A,
                                             int lda, int m0, int k0, int lane) {
  const int m  = m0 + (lane & 15);
  const int kh = (lane >> 4) * 8;
  const unsigned short* p = A + (size_t)m * (size_t)lda + (size_t)k0;
  FragU u;
  u.q[0] = *(const uint4*)(p + kh);
  u.q[1] = *(const uint4*)(p + 16 + kh);
  return u.v;
}

// B fragment: 32x16 bf16 where B[k,n] = W[n,k] (W row-major n x ldb).
// ISA layout: lane holds column n=lane%16; lanes 0-15 hold K = {k0..k0+7,
// k0+16..k0+23}; lanes 16-31 hold K = {k0+8.., k0+24..} (matches A fragment).
__device__ __forceinline__ v16bf load_b_frag(const unsigned short* __restrict__ W,
                                             int ldb, int n0, int k0, int lane) {
  const int n  = n0 + (lane & 15);
  const int kh = (lane >> 4) * 8;
  const unsigned short* p = W + (size_t)n * (size_t)ldb + (size_t)(k0 + kh);
  FragU u;
  u.q[0] = *(const uint4*)(p);
  u.q[1] = *(const uint4*)(p + 16);
  return u.v;
}

// EPI: 0 = none, 1 = relu(+bias), 2 = sigmoid
// Block = 256 threads = 8 waves tiling BM x BN wave-tiles; each wave computes
// MR stacked 16x16 tiles in M, reusing one B fragment across MR WMMAs.
template<int BM, int BN, int MR, int EPI>
__global__ void __launch_bounds__(256)
wmma_gemm_kernel(const unsigned short* __restrict__ A, int lda,
                 const unsigned short* __restrict__ B, int ldb,
                 const float* __restrict__ bias,
                 float* __restrict__ outF, int ldo, int nmax,
                 unsigned short* __restrict__ outBf, int ldbf,
                 int Kdim)
{
  const int w    = threadIdx.x >> 5;
  const int lane = threadIdx.x & 31;
  const int wm = w / BN, wn = w % BN;
  const int m0 = (blockIdx.x * BM + wm) * (16 * MR);
  const int n0 = (blockIdx.y * BN + wn) * 16;

  v8f acc[MR];
  #pragma unroll
  for (int i = 0; i < MR; ++i)
    acc[i] = (v8f){0.f, 0.f, 0.f, 0.f, 0.f, 0.f, 0.f, 0.f};

  for (int k0 = 0; k0 < Kdim; k0 += 32) {
    const v16bf b = load_b_frag(B, ldb, n0, k0, lane);
    #pragma unroll
    for (int i = 0; i < MR; ++i) {
      v16bf a = load_a_frag(A, lda, m0 + 16 * i, k0, lane);
      acc[i] = __builtin_amdgcn_wmma_f32_16x16x32_bf16(false, a, false, b,
                                                       (short)0, acc[i],
                                                       false, false);
    }
  }

  // C/D layout: VGPR r -> row base + r (lanes 0-15) or base + 8 + r (16-31),
  // column n0 + lane%16.
  const int col   = n0 + (lane & 15);
  const int rbase = (lane >> 4) * 8;
  if (col < nmax) {
    const float bv = (bias != nullptr) ? bias[col] : 0.f;
    #pragma unroll
    for (int i = 0; i < MR; ++i) {
      #pragma unroll
      for (int r = 0; r < 8; ++r) {
        const int row = m0 + 16 * i + rbase + r;
        float v = acc[i][r] + bv;
        if (EPI == 1) v = fmaxf(v, 0.f);
        if (EPI == 2) v = 1.f / (1.f + __expf(-v));
        outF[(size_t)row * (size_t)ldo + col] = v;
        if (outBf) outBf[(size_t)row * (size_t)ldbf + col] = f2bf(v);
      }
    }
  }
}

// ---------------------------------------------------------------------------
// Serial theta-base scan: h_t = relu(xin_t + W_hh h_{t-1} + b), t = 0..4095.
// 32 cooperating workgroups; each owns 32 rows of W_hh kept in LDS as bf16
// (64 KB of the 320 KB WGP LDS). Per step: read h_{t-1} (4 KB) from L2,
// 8 lanes per row reduce 1024 MACs, release/acquire barrier through L2.
// ---------------------------------------------------------------------------
__global__ void __launch_bounds__(SCAN_THREADS)
scan_kernel(const unsigned short* __restrict__ Whh_bf,
            const float* __restrict__ xin,
            const float* __restrict__ bias,
            float* __restrict__ h_out,          // phase-0 slice of d_out h area
            unsigned short* __restrict__ Hbf0,  // bf16 copy for phase GEMMs
            int* __restrict__ cnt)              // T_LEN counters, pre-zeroed
{
  extern __shared__ char smem[];
  unsigned short* sW = (unsigned short*)smem;                       // 64 KB
  float* sh = (float*)(smem + SCAN_ROWS * HID_N * sizeof(unsigned short)); // 4 KB

  const int tid = threadIdx.x;
  const int wg  = blockIdx.x;

  // Preload this WG's 32 rows of bf16 W_hh into LDS.
  const uint4* gW = (const uint4*)(Whh_bf + (size_t)wg * SCAN_ROWS * HID_N);
  uint4* sW4 = (uint4*)sW;
  for (int i = tid; i < SCAN_ROWS * HID_N / 8; i += SCAN_THREADS) sW4[i] = gW[i];

  const int r   = tid >> 3;       // row within slice (0..31)
  const int s   = tid & 7;        // k-segment (0..7), 128 elements each
  const int row = wg * SCAN_ROWS + r;
  const unsigned short* wrow = sW + r * HID_N + s * 128;

  for (int t = 0; t < T_LEN; ++t) {
    // Stage h_{t-1} into LDS (zeros at t==0). 256 float4 = 1024 floats.
    float4* sh4 = (float4*)sh;
    if (t == 0) {
      sh4[tid] = make_float4(0.f, 0.f, 0.f, 0.f);
    } else {
      const float4* hp = (const float4*)(h_out + (size_t)(t - 1) * HID_N);
      sh4[tid] = hp[tid];
    }
    __syncthreads();

    const float* hp = sh + s * 128;
    float acc = 0.f;
    #pragma unroll
    for (int k = 0; k < 128; k += 8) {
      uint4 wq = *(const uint4*)(wrow + k);
      acc = fmaf(__uint_as_float(wq.x << 16),          hp[k + 0], acc);
      acc = fmaf(__uint_as_float(wq.x & 0xFFFF0000u),  hp[k + 1], acc);
      acc = fmaf(__uint_as_float(wq.y << 16),          hp[k + 2], acc);
      acc = fmaf(__uint_as_float(wq.y & 0xFFFF0000u),  hp[k + 3], acc);
      acc = fmaf(__uint_as_float(wq.z << 16),          hp[k + 4], acc);
      acc = fmaf(__uint_as_float(wq.z & 0xFFFF0000u),  hp[k + 5], acc);
      acc = fmaf(__uint_as_float(wq.w << 16),          hp[k + 6], acc);
      acc = fmaf(__uint_as_float(wq.w & 0xFFFF0000u),  hp[k + 7], acc);
    }
    // 8-lane butterfly reduce within the wave.
    acc += __shfl_xor(acc, 1);
    acc += __shfl_xor(acc, 2);
    acc += __shfl_xor(acc, 4);

    if (s == 0) {
      float h = acc + xin[(size_t)t * HID_N + row] + bias[row];
      h = fmaxf(h, 0.f);
      h_out[(size_t)t * HID_N + row] = h;
      Hbf0[(size_t)t * HID_N + row] = f2bf(h);
    }

    // Make all stores visible at agent scope, then flag arrival for step t.
    __threadfence();
    __syncthreads();
    if (tid == 0) {
      __hip_atomic_fetch_add(&cnt[t], 1, __ATOMIC_RELEASE,
                             __HIP_MEMORY_SCOPE_AGENT);
      while (__hip_atomic_load(&cnt[t], __ATOMIC_ACQUIRE,
                               __HIP_MEMORY_SCOPE_AGENT) < SCAN_WGS)
        __builtin_amdgcn_s_sleep(1);
    }
    __syncthreads();
  }
}

// ---------------------------------------------------------------------------
// Prep / epilogue elementwise kernels
// ---------------------------------------------------------------------------
__global__ void zero_i32_kernel(int* __restrict__ p, int n) {
  int i = blockIdx.x * 256 + threadIdx.x;
  if (i < n) p[i] = 0;
}

// bf16-pack with optional row/col zero padding (src is rows_in x kin).
__global__ void pack_pad_kernel(const float* __restrict__ src,
                                unsigned short* __restrict__ dst,
                                int rows_out, int rows_in, int kout, int kin) {
  int i = blockIdx.x * 256 + threadIdx.x;
  if (i >= rows_out * kout) return;
  int rw = i / kout, k = i - rw * kout;
  float v = (rw < rows_in && k < kin) ? src[(size_t)rw * kin + k] : 0.f;
  dst[i] = f2bf(v);
}

__global__ void pack_x_kernel(const float* __restrict__ obs,
                              const float* __restrict__ act,
                              unsigned short* __restrict__ dst) {
  int i = blockIdx.x * 256 + threadIdx.x;
  if (i >= T_LEN * XK) return;
  int t = i / XK, k = i - t * XK;
  float v = 0.f;
  if (k < OBS_N)              v = obs[(size_t)t * OBS_N + k];
  else if (k < OBS_N + ACT_N) v = act[(size_t)t * ACT_N + (k - OBS_N)];
  dst[i] = f2bf(v);
}

__global__ void gather_tgt_kernel(const float* __restrict__ obs,
                                  float* __restrict__ tgt) {
  int i = blockIdx.x * 256 + threadIdx.x;
  if (i >= NPH * T_LEN * OBS_N) return;
  int j = i / (T_LEN * OBS_N);
  int rem = i - j * (T_LEN * OBS_N);
  int t = rem / OBS_N, o = rem - t * OBS_N;
  tgt[i] = obs[(size_t)(t + j) * OBS_N + o];
}

// ---------------------------------------------------------------------------
static inline size_t align256(size_t x) { return (x + 255) & ~(size_t)255; }

extern "C" void kernel_launch(void* const* d_in, const int* in_sizes, int n_in,
                              void* d_out, int out_size, void* d_ws, size_t ws_size,
                              hipStream_t stream) {
  const float* obs  = (const float*)d_in[0];  // (1, 4104, 200)
  const float* act  = (const float*)d_in[1];  // (1, 4103, 3)
  const float* Wih  = (const float*)d_in[2];  // (1024, 203)
  const float* Whh  = (const float*)d_in[3];  // (1024, 1024)
  const float* bias = (const float*)d_in[4];  // (1024,)
  const float* Wout = (const float*)d_in[5];  // (200, 1024)

  float* y_out = (float*)d_out;                               // 9*4096*200
  float* h_out = y_out + (size_t)NPH * T_LEN * OBS_N;         // 9*4096*1024
  float* tgt   = h_out + (size_t)NPH * T_LEN * HID_N;         // 9*4096*200

  // Workspace layout (~93 MB total)
  char* ws = (char*)d_ws;
  size_t off = 0;
  unsigned short* Whh_bf  = (unsigned short*)(ws + off); off = align256(off + (size_t)HID_N * HID_N * 2);
  unsigned short* Wih_bf  = (unsigned short*)(ws + off); off = align256(off + (size_t)HID_N * XK * 2);
  unsigned short* Wout_bf = (unsigned short*)(ws + off); off = align256(off + (size_t)NOUTP * HID_N * 2);
  unsigned short* xbf     = (unsigned short*)(ws + off); off = align256(off + (size_t)T_LEN * XK * 2);
  int*            cnt     = (int*)(ws + off);            off = align256(off + (size_t)T_LEN * 4);
  float*          xin     = (float*)(ws + off);          off = align256(off + (size_t)T_LEN * HID_N * 4);
  unsigned short* Hbf     = (unsigned short*)(ws + off); // 9*4096*1024 bf16

  // 0) barrier counters must be zero every call (graph replay determinism)
  zero_i32_kernel<<<(T_LEN + 255) / 256, 256, 0, stream>>>(cnt, T_LEN);

  // 1) pack weights / inputs to bf16
  pack_pad_kernel<<<(HID_N * HID_N + 255) / 256, 256, 0, stream>>>(
      Whh, Whh_bf, HID_N, HID_N, HID_N, HID_N);
  pack_pad_kernel<<<(HID_N * XK + 255) / 256, 256, 0, stream>>>(
      Wih, Wih_bf, HID_N, HID_N, XK, OBS_N + ACT_N);
  pack_pad_kernel<<<(NOUTP * HID_N + 255) / 256, 256, 0, stream>>>(
      Wout, Wout_bf, NOUTP, OBS_N, HID_N, HID_N);
  pack_x_kernel<<<(T_LEN * XK + 255) / 256, 256, 0, stream>>>(obs, act, xbf);

  // 2) xin = x @ W_ih^T : M=4096, N=1024, K=224 (WMMA, MR=4)
  wmma_gemm_kernel<1, 8, 4, 0><<<dim3(T_LEN / 64, HID_N / 128), 256, 0, stream>>>(
      xbf, XK, Wih_bf, XK, nullptr, xin, HID_N, HID_N, nullptr, 0, XK);

  // 3) serial base scan (phase 0) -> h_out phase 0 + bf16 copy
  scan_kernel<<<SCAN_WGS, SCAN_THREADS,
                SCAN_ROWS * HID_N * sizeof(unsigned short) + HID_N * sizeof(float),
                stream>>>(Whh_bf, xin, bias, h_out, Hbf, cnt);

  // 4) theta phases j=1..8: H_j = relu(H_{j-1} @ W_hh^T + b)  (WMMA, MR=4)
  for (int j = 1; j <= K_TH; ++j) {
    wmma_gemm_kernel<1, 8, 4, 1><<<dim3(T_LEN / 64, HID_N / 128), 256, 0, stream>>>(
        Hbf + (size_t)(j - 1) * T_LEN * HID_N, HID_N,
        Whh_bf, HID_N, bias,
        h_out + (size_t)j * T_LEN * HID_N, HID_N, HID_N,
        Hbf + (size_t)j * T_LEN * HID_N, HID_N, HID_N);
  }

  // 5) readout: y = sigmoid(H_all @ W_out^T), M=36864, N=208(pad)->200
  //    (WMMA, MR=4; block covers 512 rows x 16 cols)
  wmma_gemm_kernel<8, 1, 4, 2><<<dim3(NPH * T_LEN / 512, NOUTP / 16), 256, 0, stream>>>(
      Hbf, HID_N, Wout_bf, HID_N, nullptr,
      y_out, OBS_N, OBS_N, nullptr, 0, HID_N);

  // 6) obs_target gather
  gather_tgt_kernel<<<(NPH * T_LEN * OBS_N + 255) / 256, 256, 0, stream>>>(obs, tgt);
}
